// ODE_RNN_34342558499341
// MI455X (gfx1250) — matrix operand order repaired and checked by reading, not compiled, not run on hardware
//
#include <hip/hip_runtime.h>
#include <hip/hip_bf16.h>

// ---------------- problem constants ----------------
constexpr int cL   = 200;
constexpr int cB   = 512;
constexpr int cIN  = 16;
constexpr int cOUT = 8;
constexpr int cH   = 256;
constexpr int cH2  = 512;
constexpr int cG   = 1536;   // 3*H2
constexpr int cODE = 50;
constexpr int cODEP= 64;     // padded ODE hidden
constexpr int cU   = 15;
constexpr long cM  = (long)cL * cB;   // 102400 rows in time-parallel phase

typedef __bf16 bf16;
typedef __bf16 v16bf __attribute__((ext_vector_type(16)));
typedef float  v8f   __attribute__((ext_vector_type(8)));

__device__ __forceinline__ float sigm(float x) { return 1.0f / (1.0f + expf(-x)); }

// ---------------- WMMA fragment helpers (bf16 16x16x32, wave32) ----------------
// A (16x32, MxK): lanes 0-15 -> M=lane, K elems {0..7,16..23}; lanes 16-31 -> M=lane-16, K {8..15,24..31}
__device__ __forceinline__ v16bf load_a_f32(const float* __restrict__ row, int lhi) {
  int base = lhi ? 8 : 0;
  v16bf a;
#pragma unroll
  for (int i = 0; i < 8; ++i) a[i] = (bf16)row[base + i];
#pragma unroll
  for (int i = 0; i < 8; ++i) a[8 + i] = (bf16)row[base + 16 + i];
  return a;
}

__device__ __forceinline__ v16bf load_a_f32_tail(const float* __restrict__ row, int lhi, int kvalid) {
  int base = lhi ? 8 : 0;
  v16bf a;
#pragma unroll
  for (int i = 0; i < 8; ++i) { int k = base + i;      a[i]     = (bf16)((k < kvalid) ? row[k] : 0.0f); }
#pragma unroll
  for (int i = 0; i < 8; ++i) { int k = base + 16 + i; a[8 + i] = (bf16)((k < kvalid) ? row[k] : 0.0f); }
  return a;
}

// A fragment from bf16 row-major (global or LDS)
__device__ __forceinline__ v16bf load_a_bf16(const bf16* __restrict__ row, int lhi) {
  int base = lhi ? 8 : 0;
  v16bf a;
#pragma unroll
  for (int i = 0; i < 8; ++i) a[i] = row[base + i];
#pragma unroll
  for (int i = 0; i < 8; ++i) a[8 + i] = row[base + 16 + i];
  return a;
}

// B (32x16, KxN) from row-major bf16 W (Npad x Kpad), B[k][n] = W[n0+n][k0+k]:
// lanes 0-15 -> n=lane, k=k0+{0..7,16..23} ; lanes 16-31 -> n=lane-16, k=k0+{8..15,24..31}
__device__ __forceinline__ v16bf load_b(const bf16* __restrict__ Wb, int ldw, int n0, int k0, int lane) {
  const bf16* p = Wb + (long)(n0 + (lane & 15)) * ldw + k0 + ((lane & 16) ? 8 : 0);
  v16bf b;
#pragma unroll
  for (int i = 0; i < 8; ++i) b[i] = p[i];
#pragma unroll
  for (int i = 0; i < 8; ++i) b[8 + i] = p[16 + i];
  return b;
}

__device__ __forceinline__ v8f wmma_bf16(v16bf a, v16bf b, v8f c) {
  return __builtin_amdgcn_wmma_f32_16x16x32_bf16(false, a, false, b, (short)0, c, false, false);
}

// ---------------- weight conversion: fp32 (N,K) -> bf16 (Npad,Kpad), zero padded ----------------
__global__ void cvt_w_kernel(const float* __restrict__ W, bf16* __restrict__ Wb,
                             int N, int K, int Npad, int Kpad) {
  int i = blockIdx.x * blockDim.x + threadIdx.x;
  int total = Npad * Kpad;
  if (i >= total) return;
  int n = i / Kpad, k = i % Kpad;
  float v = (n < N && k < K) ? W[(long)n * K + k] : 0.0f;
  Wb[i] = (bf16)v;
}

// ---------------- generic GEMM: Y = act(X @ Wb^T + bias) ----------------
// each wave: one 16-row M tile x 4 consecutive N tiles (A fragment shared across the 4 tiles)
__global__ __launch_bounds__(128)
void gemm_bias_act_kernel(const float* __restrict__ X, int ldx,
                          const bf16* __restrict__ Wb, int ldw,
                          const float* __restrict__ bias,
                          float* __restrict__ Y, int ldy,
                          int Mtiles, int Ngroups,          // Ngroups = Ntiles/4
                          int kfull, int ktail, int act) {
  int lane = threadIdx.x & 31;
  int wid  = threadIdx.x >> 5;
  int grp  = blockIdx.x * 4 + wid;
  if (grp >= Mtiles * Ngroups) return;
  int mt = grp / Ngroups, ng = grp % Ngroups;
  int m0 = mt * 16, n0b = ng * 64;
  int lhi = (lane & 16) ? 1 : 0;
  const float* arow = X + (long)(m0 + (lane & 15)) * ldx;
  v8f c[4] = {};
  for (int k0 = 0; k0 < kfull; k0 += 32) {
    v16bf a = load_a_f32(arow + k0, lhi);
#pragma unroll
    for (int j = 0; j < 4; ++j) {
      v16bf b = load_b(Wb, ldw, n0b + j * 16, k0, lane);
      c[j] = wmma_bf16(a, b, c[j]);
    }
  }
  if (ktail > 0) {
    v16bf a = load_a_f32_tail(arow + kfull, lhi, ktail);
#pragma unroll
    for (int j = 0; j < 4; ++j) {
      v16bf b = load_b(Wb, ldw, n0b + j * 16, kfull, lane);
      c[j] = wmma_bf16(a, b, c[j]);
    }
  }
#pragma unroll
  for (int j = 0; j < 4; ++j) {
    int col = n0b + j * 16 + (lane & 15);
    float bv = bias ? bias[col] : 0.0f;
#pragma unroll
    for (int v = 0; v < 8; ++v) {
      int row = m0 + v + (lhi ? 8 : 0);
      float val = c[j][v] + bv;
      if (act == 1) val = tanhf(val);
      Y[(long)row * ldy + col] = val;
    }
  }
}

// ---------------- fused dblock + heads ----------------
// one 16-row block per workgroup (8 waves). Stage 1: t = tanh(.)*sigm(.) -> LDS (bf16).
// Stage 2: wave 0 computes mu/ls heads (K=512) reading A fragments from LDS.
__global__ __launch_bounds__(256)
void dblock_heads_kernel(const float* __restrict__ hn,
                         const bf16* __restrict__ w1b, const bf16* __restrict__ w2b,
                         const float* __restrict__ b1, const float* __restrict__ b2,
                         const bf16* __restrict__ wmub, const bf16* __restrict__ wlsb,
                         const float* __restrict__ bmu, const float* __restrict__ bls,
                         const float* __restrict__ noise_t,
                         float* __restrict__ mu_out, float* __restrict__ ls_out,
                         float* __restrict__ x_out) {
  __shared__ __align__(16) bf16 tS[16 * cH2];   // 16 KB
  int lane = threadIdx.x & 31, wid = threadIdx.x >> 5;
  int m0 = blockIdx.x * 16;
  int lhi = (lane & 16) ? 1 : 0;
  const float* arow = hn + (long)(m0 + (lane & 15)) * cH2;
  v8f c1[4] = {};
  v8f c2[4] = {};
  for (int k0 = 0; k0 < cH2; k0 += 32) {
    v16bf a = load_a_f32(arow + k0, lhi);
#pragma unroll
    for (int j = 0; j < 4; ++j) {
      int n0 = (wid * 4 + j) * 16;
      v16bf b = load_b(w1b, cH2, n0, k0, lane);
      c1[j] = wmma_bf16(a, b, c1[j]);
      b = load_b(w2b, cH2, n0, k0, lane);
      c2[j] = wmma_bf16(a, b, c2[j]);
    }
  }
#pragma unroll
  for (int j = 0; j < 4; ++j) {
    int col = (wid * 4 + j) * 16 + (lane & 15);
    float bb1 = b1[col], bb2 = b2[col];
#pragma unroll
    for (int v = 0; v < 8; ++v) {
      int row = v + (lhi ? 8 : 0);
      tS[row * cH2 + col] = (bf16)(tanhf(c1[j][v] + bb1) * sigm(c2[j][v] + bb2));
    }
  }
  __syncthreads();
  if (wid == 0) {
    const bf16* als = tS + (lane & 15) * cH2;
    v8f d1 = {}, d2 = {};
    for (int k0 = 0; k0 < cH2; k0 += 32) {
      v16bf a = load_a_bf16(als + k0, lhi);
      v16bf b = load_b(wmub, cH2, 0, k0, lane);
      d1 = wmma_bf16(a, b, d1);
      b = load_b(wlsb, cH2, 0, k0, lane);
      d2 = wmma_bf16(a, b, d2);
    }
    int col = lane & 15;
    if (col < cOUT) {
      float bm = bmu[col], bl = bls[col];
#pragma unroll
      for (int v = 0; v < 8; ++v) {
        int row = m0 + v + (lhi ? 8 : 0);
        float mu = d1[v] + bm;
        float ls = d2[v] + bl;
        float eps = noise_t[(long)row * cOUT + col];
        mu_out[(long)row * cOUT + col] = mu;
        ls_out[(long)row * cOUT + col] = ls;
        x_out[(long)row * cOUT + col]  = mu + expf(0.5f * ls) * eps;
      }
    }
  }
}

// ---------------- fused GRU (gi + gh + gates) + h_seq writeback ----------------
__global__ __launch_bounds__(128)
void gru_kernel(const float* __restrict__ u_emb_t,   // (B, 256)
                const float* __restrict__ x_emb_t,   // (B, 256)
                const float* __restrict__ hn_old,    // (B, 512)
                const bf16* __restrict__ wihb,       // (1536, 512) bf16
                const bf16* __restrict__ whhb,       // (1536, 512) bf16
                const float* __restrict__ bih, const float* __restrict__ bhh,
                float* __restrict__ hn_new,
                float* __restrict__ hseq_t) {        // h_seq[t] = hn_old
  int lane = threadIdx.x & 31, wid = threadIdx.x >> 5;
  int tile = blockIdx.x * 4 + wid;            // 1024 tiles = 32x32
  int mt = tile >> 5, nt = tile & 31;
  int m0 = mt * 16, n0 = nt * 16;
  int lhi = (lane & 16) ? 1 : 0;
  int mr = m0 + (lane & 15);
  const float* hrow = hn_old  + (long)mr * cH2;
  const float* urow = u_emb_t + (long)mr * cH;
  const float* xrow = x_emb_t + (long)mr * cH;
  v8f ir = {}, iz = {}, inn = {}, hr = {}, hz = {}, hnv = {};
  for (int k0 = 0; k0 < cH2; k0 += 32) {
    const float* xs = (k0 < cH) ? (urow + k0) : (xrow + (k0 - cH));
    v16bf ax = load_a_f32(xs, lhi);
    v16bf ah = load_a_f32(hrow + k0, lhi);
    v16bf b;
    b = load_b(wihb, cH2, n0,            k0, lane); ir  = wmma_bf16(ax, b, ir);
    b = load_b(wihb, cH2, cH2 + n0,      k0, lane); iz  = wmma_bf16(ax, b, iz);
    b = load_b(wihb, cH2, 2 * cH2 + n0,  k0, lane); inn = wmma_bf16(ax, b, inn);
    b = load_b(whhb, cH2, n0,            k0, lane); hr  = wmma_bf16(ah, b, hr);
    b = load_b(whhb, cH2, cH2 + n0,      k0, lane); hz  = wmma_bf16(ah, b, hz);
    b = load_b(whhb, cH2, 2 * cH2 + n0,  k0, lane); hnv = wmma_bf16(ah, b, hnv);
  }
  int col = n0 + (lane & 15);
  float bihr = bih[col], bihz = bih[cH2 + col], bihn = bih[2 * cH2 + col];
  float bhhr = bhh[col], bhhz = bhh[cH2 + col], bhhn = bhh[2 * cH2 + col];
#pragma unroll
  for (int v = 0; v < 8; ++v) {
    int row = m0 + v + (lhi ? 8 : 0);
    long idx = (long)row * cH2 + col;
    float r = sigm(ir[v] + bihr + hr[v] + bhhr);
    float z = sigm(iz[v] + bihz + hz[v] + bhhz);
    float n = tanhf(inn[v] + bihn + r * (hnv[v] + bhhn));
    float hp = hn_old[idx];
    hseq_t[idx] = hp;                         // pre-update carry -> h_seq[t]
    hn_new[idx] = (1.0f - z) * n + z * hp;
  }
}

// ---------------- fused ODE Euler step: a = tanh(h@w1^T+b1); h += dt*(a@w2^T+b2) ----------------
// one 16-row block per workgroup (4 waves); `a` staged as bf16 in LDS.
__global__ __launch_bounds__(128)
void ode_kernel(float* __restrict__ hn_new,          // (B, 512), h = cols 0..255 (in place)
                const bf16* __restrict__ w1b,        // (64, 256) bf16
                const float* __restrict__ b1,
                const bf16* __restrict__ w2b,        // (256, 64) bf16 (k 50..63 zero)
                const float* __restrict__ b2,
                const float* __restrict__ dt_src) {  // ext_input + t*B*16, dt at col 15
  __shared__ __align__(16) bf16 aS[16 * cODEP];      // 2 KB
  int lane = threadIdx.x & 31, wid = threadIdx.x >> 5;
  int m0 = blockIdx.x * 16;
  int lhi = (lane & 16) ? 1 : 0;
  // stage 1: wave `wid` computes n-tile `wid` of a (N=64)
  {
    const float* arow = hn_new + (long)(m0 + (lane & 15)) * cH2;
    v8f c = {};
    for (int k0 = 0; k0 < cH; k0 += 32) {
      v16bf a = load_a_f32(arow + k0, lhi);
      v16bf b = load_b(w1b, cH, wid * 16, k0, lane);
      c = wmma_bf16(a, b, c);
    }
    int col = wid * 16 + (lane & 15);
    float bb = (col < cODE) ? b1[col] : 0.0f;
#pragma unroll
    for (int v = 0; v < 8; ++v) {
      int row = v + (lhi ? 8 : 0);
      float val = (col < cODE) ? tanhf(c[v] + bb) : 0.0f;
      aS[row * cODEP + col] = (bf16)val;
    }
  }
  __syncthreads();
  // stage 2: 16 n-tiles (N=256) over 4 waves, K=64 from LDS
  const bf16* als = aS + (lane & 15) * cODEP;
#pragma unroll
  for (int j = 0; j < 4; ++j) {
    int n0 = (wid * 4 + j) * 16;
    v8f c = {};
#pragma unroll
    for (int k0 = 0; k0 < cODEP; k0 += 32) {
      v16bf a = load_a_bf16(als + k0, lhi);
      v16bf b = load_b(w2b, cODEP, n0, k0, lane);
      c = wmma_bf16(a, b, c);
    }
    int col = n0 + (lane & 15);
    float bb2 = b2[col];
#pragma unroll
    for (int v = 0; v < 8; ++v) {
      int row = m0 + v + (lhi ? 8 : 0);
      float dt = dt_src[(long)row * cIN + (cIN - 1)];
      long idx = (long)row * cH2 + col;
      hn_new[idx] = hn_new[idx] + dt * (c[v] + bb2);
    }
  }
}

// ---------------- host launcher ----------------
extern "C" void kernel_launch(void* const* d_in, const int* in_sizes, int n_in,
                              void* d_out, int out_size, void* d_ws, size_t ws_size,
                              hipStream_t stream) {
  const float* ext    = (const float*)d_in[0];   // (L,B,16)
  const float* obs    = (const float*)d_in[1];   // (L,B,8)
  const float* noise  = (const float*)d_in[2];   // (L,B,8)
  const float* pu_w1  = (const float*)d_in[3];
  const float* pu_b1  = (const float*)d_in[4];
  const float* pu_w2  = (const float*)d_in[5];
  const float* pu_b2  = (const float*)d_in[6];
  const float* px_w1  = (const float*)d_in[7];
  const float* px_b1  = (const float*)d_in[8];
  const float* px_w2  = (const float*)d_in[9];
  const float* px_b2  = (const float*)d_in[10];
  const float* gwih   = (const float*)d_in[11];
  const float* gwhh   = (const float*)d_in[12];
  const float* gbih   = (const float*)d_in[13];
  const float* gbhh   = (const float*)d_in[14];
  const float* ly_w1  = (const float*)d_in[15];
  const float* ly_b1  = (const float*)d_in[16];
  const float* ly_w2  = (const float*)d_in[17];
  const float* ly_b2  = (const float*)d_in[18];
  const float* ly_wmu = (const float*)d_in[19];
  const float* ly_bmu = (const float*)d_in[20];
  const float* ly_wls = (const float*)d_in[21];
  const float* ly_bls = (const float*)d_in[22];
  const float* ode_w1 = (const float*)d_in[23];
  const float* ode_b1 = (const float*)d_in[24];
  const float* ode_w2 = (const float*)d_in[25];
  const float* ode_b2 = (const float*)d_in[26];

  float* out = (float*)d_out;
  const long SZ8 = (long)cL * cB * cOUT;     // 819200
  float* o_mu    = out;
  float* o_ls    = out + SZ8;
  float* o_hseq  = out + 2 * SZ8;
  float* o_xt    = out + 2 * SZ8 + (long)cL * cB * cH2;
  float* o_xemb  = o_xt + SZ8;
  float* o_hnfin = o_xemb + (long)cL * cB * cH;

  // ---- carve workspace ----
  char* wp = (char*)d_ws;
  auto carve = [&](size_t bytes) -> char* {
    char* p = wp;
    wp += (bytes + 255) & ~(size_t)255;
    return p;
  };
  float* bufA   = (float*)carve(sizeof(float) * cM * cH);   // 105 MB intermediate
  float* u_emb  = (float*)carve(sizeof(float) * cM * cH);   // 105 MB u_embed
  bf16* pu_w1b  = (bf16*)carve(2 * 256 * 32);
  bf16* pu_w2b  = (bf16*)carve(2 * 256 * 256);
  bf16* px_w1b  = (bf16*)carve(2 * 256 * 32);
  bf16* px_w2b  = (bf16*)carve(2 * 256 * 256);
  bf16* gwihb   = (bf16*)carve(2 * (size_t)cG * cH2);
  bf16* gwhhb   = (bf16*)carve(2 * (size_t)cG * cH2);
  bf16* ly_w1b  = (bf16*)carve(2 * (size_t)cH2 * cH2);
  bf16* ly_w2b  = (bf16*)carve(2 * (size_t)cH2 * cH2);
  bf16* ly_wmub = (bf16*)carve(2 * 16 * cH2);
  bf16* ly_wlsb = (bf16*)carve(2 * 16 * cH2);
  bf16* ode_w1b = (bf16*)carve(2 * cODEP * cH);
  bf16* ode_w2b = (bf16*)carve(2 * cH * cODEP);
  float* hn0    = (float*)carve(sizeof(float) * cB * cH2);
  float* hn1    = (float*)carve(sizeof(float) * cB * cH2);

  // ---- weight conversions (fp32 -> padded bf16) ----
  auto cvt = [&](const float* W, bf16* Wb, int N, int K, int Npad, int Kpad) {
    int total = Npad * Kpad;
    cvt_w_kernel<<<(total + 255) / 256, 256, 0, stream>>>(W, Wb, N, K, Npad, Kpad);
  };
  cvt(pu_w1,  pu_w1b,  256, cU,   256, 32);
  cvt(pu_w2,  pu_w2b,  256, 256,  256, 256);
  cvt(px_w1,  px_w1b,  256, cOUT, 256, 32);
  cvt(px_w2,  px_w2b,  256, 256,  256, 256);
  cvt(gwih,   gwihb,   cG,  cH2,  cG,  cH2);
  cvt(gwhh,   gwhhb,   cG,  cH2,  cG,  cH2);
  cvt(ly_w1,  ly_w1b,  cH2, cH2,  cH2, cH2);
  cvt(ly_w2,  ly_w2b,  cH2, cH2,  cH2, cH2);
  cvt(ly_wmu, ly_wmub, cOUT, cH2, 16,  cH2);
  cvt(ly_wls, ly_wlsb, cOUT, cH2, 16,  cH2);
  cvt(ode_w1, ode_w1b, cODE, cH,  cODEP, cH);
  cvt(ode_w2, ode_w2b, cH,  cODE, cH,  cODEP);

  // ---- phase A: time-parallel embeddings (M = 102400) ----
  const int MT = cM / 16;                     // 6400 M tiles
  const int NG = (cH / 16) / 4;               // 4 groups of 4 N-tiles
  const int blksA = (MT * NG) / 4;            // 6400 blocks of 4 waves
  // bufA = tanh(u_seq @ pu_w1^T + b1)   (u = cols 0..14 of ext, row stride 16)
  gemm_bias_act_kernel<<<blksA, 128, 0, stream>>>(ext, cIN, pu_w1b, 32, pu_b1,
                                                  bufA, cH, MT, NG, 0, cU, 1);
  // u_emb = tanh(bufA @ pu_w2^T + b2)
  gemm_bias_act_kernel<<<blksA, 128, 0, stream>>>(bufA, cH, pu_w2b, 256, pu_b2,
                                                  u_emb, cH, MT, NG, 256, 0, 1);
  // bufA = tanh(obs @ px_w1^T + b1)
  gemm_bias_act_kernel<<<blksA, 128, 0, stream>>>(obs, cOUT, px_w1b, 32, px_b1,
                                                  bufA, cH, MT, NG, 0, cOUT, 1);
  // x_embed (output) = tanh(bufA @ px_w2^T + b2)
  gemm_bias_act_kernel<<<blksA, 128, 0, stream>>>(bufA, cH, px_w2b, 256, px_b2,
                                                  o_xemb, cH, MT, NG, 256, 0, 1);

  // ---- sequential scan: 3 fused kernels per step ----
  hipMemsetAsync(hn0, 0, sizeof(float) * cB * cH2, stream);
  float* hbuf[2] = { hn0, hn1 };
  for (int t = 0; t < cL; ++t) {
    float* hc  = hbuf[t & 1];
    float* hnx = hbuf[(t + 1) & 1];
    dblock_heads_kernel<<<cB / 16, 256, 0, stream>>>(hc, ly_w1b, ly_w2b, ly_b1, ly_b2,
                                                     ly_wmub, ly_wlsb, ly_bmu, ly_bls,
                                                     noise + (long)t * cB * cOUT,
                                                     o_mu + (long)t * cB * cOUT,
                                                     o_ls + (long)t * cB * cOUT,
                                                     o_xt + (long)t * cB * cOUT);
    gru_kernel<<<256, 128, 0, stream>>>(u_emb + (long)t * cB * cH,
                                        o_xemb + (long)t * cB * cH,
                                        hc, gwihb, gwhhb, gbih, gbhh, hnx,
                                        o_hseq + (long)t * cB * cH2);
    ode_kernel<<<cB / 16, 128, 0, stream>>>(hnx, ode_w1b, ode_b1, ode_w2b, ode_b2,
                                            ext + (long)t * cB * cIN);
  }
  // final carry: after step 199, state is in hbuf[0] (L even)
  hipMemcpyAsync(o_hnfin, hbuf[cL & 1], sizeof(float) * cB * cH2,
                 hipMemcpyDeviceToDevice, stream);
}
